// MultiHeadLatentAttention_58222576665016
// MI455X (gfx1250) — matrix operand order repaired and checked by reading, not compiled, not run on hardware
//
#include <hip/hip_runtime.h>
#include <hip/hip_bf16.h>

typedef unsigned short u16;
typedef unsigned int   u32;
typedef __attribute__((ext_vector_type(16))) __bf16 v16bf;
typedef __attribute__((ext_vector_type(8)))  float  v8f;
typedef __attribute__((ext_vector_type(4)))  u32    v4u;
typedef __attribute__((ext_vector_type(2)))  u32    v2u;

union FragAB { v16bf bf; u16 u[16]; u32 d[8]; v4u q[2]; };
union FragC  { v8f v; float f[8]; };

__device__ __forceinline__ u16 f32_to_bf16(float f) {
  u32 x = __float_as_uint(f);
  u32 r = x + 0x7fffu + ((x >> 16) & 1u);
  if ((x & 0x7f800000u) == 0x7f800000u) r = x;  // preserve inf/nan
  return (u16)(r >> 16);
}
__device__ __forceinline__ float bf16_to_f32(u16 h) {
  return __uint_as_float(((u32)h) << 16);
}

// --------------------------------------------------------- async LDS copy ---
// CDNA5 async global->LDS (tracked by ASYNCcnt). lds_addr is a byte offset in
// the workgroup LDS allocation (low 32 bits of the generic pointer).
__device__ __forceinline__ void async_copy_b128(u32 lds_addr, const void* gaddr) {
  asm volatile("global_load_async_to_lds_b128 %0, %1, off"
               :: "v"(lds_addr), "v"(gaddr) : "memory");
}
__device__ __forceinline__ void wait_async0() {
  asm volatile("s_wait_asynccnt 0x0" ::: "memory");
}
__device__ __forceinline__ u32 lds_off(const void* p) {
  return (u32)(size_t)p;
}

// ---------------------------------------------------------------- convert ---
__global__ void cvt_f32_bf16(const float* __restrict__ in, u16* __restrict__ out, int n) {
  int i = blockIdx.x * blockDim.x + threadIdx.x;
  if (i < n) out[i] = f32_to_bf16(in[i]);
}

// --------------------------------------------------------- WMMA frag loads ---
// A matrix 16x32 bf16 from row-major source, leading dim ld (elements).
// Per-lane: 2 contiguous 16B chunks (K groups {kb..kb+7}, {16+kb..16+kb+7}).
__device__ __forceinline__ FragAB load_a_rm(const u16* base, int ld) {
  FragAB f;
  int lane = threadIdx.x & 31;
  const u16* p = base + (lane & 15) * ld + ((lane < 16) ? 0 : 8);
  f.q[0] = *(const v4u*)(p);
  f.q[1] = *(const v4u*)(p + 16);
  return f;
}
// B fragment from fragment-major LDS: Bf[...subtile][lane][8 dwords].
// base_dw = dword offset of the subtile block (subtile_index * 256).
__device__ __forceinline__ FragAB load_b_fm(const u32* Bf, int base_dw) {
  FragAB f;
  int lane = threadIdx.x & 31;
  const u32* p = Bf + base_dw + lane * 8;
  f.q[0] = *(const v4u*)(p);
  f.q[1] = *(const v4u*)(p + 4);
  return f;
}
// pack two global rows (k, k+1) of bf16 into K-pair dwords (for cols n0, n0+1)
__device__ __forceinline__ v2u pack_pair(u32 g0, u32 g1) {
  v2u r;
  r.x = (g0 & 0xffffu) | (g1 << 16);
  r.y = (g0 >> 16) | (g1 & 0xffff0000u);
  return r;
}
// scatter a packed K-pair into fragment-major storage.
// kp = k/2 within the 32-K chunk is kpl (0..15); n0 even.
__device__ __forceinline__ void store_b_fm(u32* Bf, int chunk_base_dw, int kpl, int n0,
                                           v2u r) {
  int st   = n0 >> 4;
  int lane = (((kpl >> 2) & 1) ? 16 : 0) + (n0 & 15);
  int i    = (kpl & 3) + ((kpl & 8) ? 4 : 0);
  int d0   = chunk_base_dw + st * 256 + lane * 8 + i;
  Bf[d0]     = r.x;   // col n0
  Bf[d0 + 8] = r.y;   // col n0+1 (next lane)
}

// ------------------------------------------------------------------- GEMM ---
// C[M,N] = A[M,K] @ B[K,N], bf16 in, f32 accumulate.
// Epilogue column remap: col = (cgroup==0) ? n : (n/cgroup)*192 + coff + n%cgroup
template <bool OUT_BF16>
__global__ __launch_bounds__(256) void gemm_bf16(
    const u16* __restrict__ A, const u16* __restrict__ B, void* __restrict__ C,
    int M, int N, int K, int ldc, int cgroup, int coff) {
  __shared__ __align__(16) u16 As[64 * 32];
  __shared__ __align__(16) u32 Bf[4 * 32 * 8];  // fragment-major: [nsub][lane][8]

  int tid  = threadIdx.x;
  int lane = tid & 31;
  int wave = tid >> 5;
  int wm = wave & 3;   // 4 waves along M (16 rows each)
  int wn = wave >> 2;  // 2 waves along N (32 cols each)
  int bm = blockIdx.y * 64;
  int bn = blockIdx.x * 64;

  FragC acc[2];
#pragma unroll
  for (int t = 0; t < 2; ++t)
#pragma unroll
    for (int r = 0; r < 8; ++r) acc[t].f[r] = 0.f;

  u32 as_base = lds_off(As);

  for (int k0 = 0; k0 < K; k0 += 32) {
    __syncthreads();
    // A tile 64x32 halves = 256 x b128 chunks : 1 per thread, async DMA.
    {
      int row = tid >> 2, qo = tid & 3;  // qo in 16B units within the 64B row
      async_copy_b128(as_base + row * 64 + qo * 16,
                      A + (size_t)(bm + row) * K + k0 + qo * 8);
    }
    // B tile 32x64 -> fragment-major Bf : 512 dword pairs, 2 per thread.
    for (int idx = tid; idx < 512; idx += 256) {
      int kp = idx >> 5, np = idx & 31;
      int n0 = np * 2;
      u32 g0 = *(const u32*)(B + (size_t)(k0 + 2 * kp) * N + bn + n0);
      u32 g1 = *(const u32*)(B + (size_t)(k0 + 2 * kp + 1) * N + bn + n0);
      store_b_fm(Bf, 0, kp, n0, pack_pair(g0, g1));
    }
    // prefetch next B panel (gfx1250 global_prefetch)
    if (k0 + 32 < K) __builtin_prefetch(B + (size_t)(k0 + 32 + (tid >> 3)) * N + bn, 0, 1);
    wait_async0();
    __syncthreads();

    FragAB a = load_a_rm(As + wm * 16 * 32, 32);
#pragma unroll
    for (int t = 0; t < 2; ++t) {
      FragAB b = load_b_fm(Bf, (wn * 2 + t) * 256);
      acc[t].v = __builtin_amdgcn_wmma_f32_16x16x32_bf16(
          false, a.bf, false, b.bf, (short)0, acc[t].v, false, false);
    }
  }

  int nn = lane & 15;
  int mo = (lane >> 4) << 3;
#pragma unroll
  for (int t = 0; t < 2; ++t) {
#pragma unroll
    for (int r = 0; r < 8; ++r) {
      int m = bm + wm * 16 + r + mo;
      int n = bn + wn * 32 + t * 16 + nn;
      int col = (cgroup == 0) ? n : (n / cgroup) * 192 + coff + (n % cgroup);
      if (OUT_BF16)
        ((u16*)C)[(size_t)m * ldc + col] = f32_to_bf16(acc[t].f[r]);
      else
        ((float*)C)[(size_t)m * ldc + col] = acc[t].f[r];
    }
  }
}

// ------------------------------------------------------------------- RoPE ---
// In-place rotary on cols [128,192) of rows laid out [B*S*H, 192].
__global__ void rope_inplace(u16* __restrict__ buf, const float* __restrict__ cosb,
                             const float* __restrict__ sinb, int S, int H) {
  int idx = blockIdx.x;  // (b*S + s)*H + h
  int s = (idx / H) % S;
  u16* row = buf + (size_t)idx * 192 + 128;
  int j = threadIdx.x;
  if (j < 32) {
    float xr = bf16_to_f32(row[2 * j]);
    float xi = bf16_to_f32(row[2 * j + 1]);
    float c = cosb[s * 32 + j];
    float sn = sinb[s * 32 + j];
    row[2 * j]     = f32_to_bf16(xr * c - xi * sn);
    row[2 * j + 1] = f32_to_bf16(xr * sn + xi * c);
  }
}

// -------------------------------------------------------- flash attention ---
// One block per (b, h, 64-row q tile). d_qk = 192, d_v = 128, causal.
__global__ __launch_bounds__(256) void mla_attention(
    const u16* __restrict__ Q, const u16* __restrict__ Kb, const u16* __restrict__ Vb,
    u16* __restrict__ O, int Slen) {
  __shared__ __align__(16) u16 Klds[64 * 192];    // keys row-major (async DMA)
  __shared__ __align__(16) u32 Vf[2 * 8 * 32 * 8];  // V fragment-major: [kc][nsub][lane][8]
  __shared__ __align__(16) float Sbuf[64 * 64];   // P(bf16) aliased into low half of each row
  __shared__ float mrowS[64], lrowS[64], arowS[64];

  int qt = blockIdx.x & 31;
  int h  = (blockIdx.x >> 5) & 15;
  int b  = blockIdx.x >> 9;
  int q0 = qt * 64;
  int tid  = threadIdx.x;
  int lane = tid & 31;
  int wave = tid >> 5;
  int wm = wave & 3;
  int wn = wave >> 2;
  const float scale = 0.08838834764831845f;  // 1/sqrt(128) per reference

  // Q fragments (6 chunks of K=32 covering d=192), held in registers.
  FragAB qf[6];
  {
    int m  = lane & 15;
    int kb = (lane < 16) ? 0 : 8;
    int qrow = q0 + wm * 16 + m;
    const u16* qp = Q + ((size_t)(b * Slen + qrow) * 16 + h) * 192 + kb;
#pragma unroll
    for (int f = 0; f < 6; ++f) {
      qf[f].q[0] = *(const v4u*)(qp + f * 32);
      qf[f].q[1] = *(const v4u*)(qp + f * 32 + 16);
    }
  }
  if (tid < 64) { mrowS[tid] = -1e30f; lrowS[tid] = 0.f; }

  FragC oacc[4];
#pragma unroll
  for (int t = 0; t < 4; ++t)
#pragma unroll
    for (int r = 0; r < 8; ++r) oacc[t].f[r] = 0.f;

  u32 klds_base = lds_off(Klds);

  for (int jt = 0; jt <= qt; ++jt) {
    int k0 = jt * 64;
    __syncthreads();
    // K tile 64x192 halves = 1536 x b128 chunks : 6 per thread, async DMA.
#pragma unroll
    for (int c = 0; c < 6; ++c) {
      int idx = c * 256 + tid;
      int r = idx / 24, qo = idx % 24;  // 24 x 16B per 384B row
      async_copy_b128(klds_base + r * 384 + qo * 16,
                      Kb + ((size_t)(b * Slen + k0 + r) * 16 + h) * 192 + qo * 8);
    }
    // V tile 64x128 -> fragment-major Vf : 2048 dword pairs, 8 per thread.
    for (int idx = tid; idx < 2048; idx += 256) {
      int kpg = idx >> 6, np = idx & 63;  // kpg = k/2 (0..31), col pair np
      int n0 = np * 2;
      u32 g0 = *(const u32*)(Vb + ((size_t)(b * Slen + k0 + 2 * kpg) * 16 + h) * 128 + n0);
      u32 g1 = *(const u32*)(Vb + ((size_t)(b * Slen + k0 + 2 * kpg + 1) * 16 + h) * 128 + n0);
      store_b_fm(Vf, (kpg >> 4) * (8 * 256), kpg & 15, n0, pack_pair(g0, g1));
    }
    wait_async0();
    __syncthreads();

    // S = Q @ K^T  (each wave: 16 rows x 32 cols = two 16x16 tiles)
#pragma unroll
    for (int t = 0; t < 2; ++t) {
      FragC sc;
#pragma unroll
      for (int r = 0; r < 8; ++r) sc.f[r] = 0.f;
      int key = wn * 32 + t * 16 + (lane & 15);
      int kb2 = (lane < 16) ? 0 : 8;
      const u16* kp_ = Klds + key * 192 + kb2;
#pragma unroll
      for (int f = 0; f < 6; ++f) {
        FragAB bt;  // B[c][n] = K[key][c]; two 8-half chunks per lane (kb2, 16+kb2)
        bt.q[0] = *(const v4u*)(kp_ + f * 32);
        bt.q[1] = *(const v4u*)(kp_ + f * 32 + 16);
        sc.v = __builtin_amdgcn_wmma_f32_16x16x32_bf16(
            false, qf[f].bf, false, bt.bf, (short)0, sc.v, false, false);
      }
      int nn = lane & 15;
      int mo = (lane >> 4) << 3;
#pragma unroll
      for (int r = 0; r < 8; ++r) {
        int ml = wm * 16 + r + mo;
        int nc = wn * 32 + t * 16 + nn;
        float s = sc.f[r] * scale;
        if (k0 + nc > q0 + ml) s = -1e30f;  // causal (only diagonal tile triggers)
        Sbuf[ml * 64 + nc] = s;
      }
    }
    __syncthreads();

    // Online softmax; P(bf16) aliased into low 128B of each 256B S row.
    if (tid < 64) {
      int row = tid;
      volatile float* srow = &Sbuf[row * 64];
      float mx = -1e30f;
      for (int c = 0; c < 64; ++c) { float v = srow[c]; mx = v > mx ? v : mx; }
      float mold = mrowS[row];
      float mnew = mx > mold ? mx : mold;
      float alpha = __expf(mold - mnew);
      volatile u16* prow = (volatile u16*)&Sbuf[row * 64];
      float sum = 0.f;
      for (int c = 0; c < 64; ++c) {  // ascending c: P write idx c/2 <= already-read c
        float p = __expf(srow[c] - mnew);
        sum += p;
        prow[c] = f32_to_bf16(p);
      }
      mrowS[row] = mnew;
      lrowS[row] = lrowS[row] * alpha + sum;
      arowS[row] = alpha;
    }
    __syncthreads();

    // O = O*alpha + P @ V  (each wave: 16 rows x 64 cols = four 16x16 tiles)
    {
      int mo = (lane >> 4) << 3;
#pragma unroll
      for (int t = 0; t < 4; ++t)
#pragma unroll
        for (int r = 0; r < 8; ++r) oacc[t].f[r] *= arowS[wm * 16 + r + mo];

#pragma unroll
      for (int kc = 0; kc < 2; ++kc) {
        FragAB pa;  // P as A-matrix, row pitch 128 u16 (aliased S rows)
        {
          int m  = lane & 15;
          int kb = (lane < 16) ? 0 : 8;
          const u16* pb = (const u16*)Sbuf + (size_t)(wm * 16 + m) * 128 + kc * 32 + kb;
          pa.q[0] = *(const v4u*)(pb);
          pa.q[1] = *(const v4u*)(pb + 16);
        }
#pragma unroll
        for (int t = 0; t < 4; ++t) {
          FragAB vb = load_b_fm(Vf, (kc * 8 + wn * 4 + t) * 256);
          oacc[t].v = __builtin_amdgcn_wmma_f32_16x16x32_bf16(
              false, pa.bf, false, vb.bf, (short)0, oacc[t].v, false, false);
        }
      }
    }
  }
  __syncthreads();

  // Normalize and store attn[b, q, h*128 + c] (bf16).
  {
    int nn = lane & 15;
    int mo = (lane >> 4) << 3;
#pragma unroll
    for (int t = 0; t < 4; ++t)
#pragma unroll
      for (int r = 0; r < 8; ++r) {
        int ml = wm * 16 + r + mo;
        float inv = 1.0f / lrowS[ml];
        int col = h * 128 + wn * 64 + t * 16 + nn;
        O[(size_t)(b * Slen + q0 + ml) * 2048 + col] = f32_to_bf16(oacc[t].f[r] * inv);
      }
  }
}

// ----------------------------------------------------------------- driver ---
extern "C" void kernel_launch(void* const* d_in, const int* in_sizes, int n_in,
                              void* d_out, int out_size, void* d_ws, size_t ws_size,
                              hipStream_t stream) {
  (void)n_in; (void)out_size; (void)ws_size; (void)in_sizes;
  constexpr int B = 2, S = 2048, DM = 2048, H = 16, DC = 512;
  constexpr int M = B * S;  // 4096 tokens

  const float* x    = (const float*)d_in[0];
  const float* fcos = (const float*)d_in[1];
  const float* fsin = (const float*)d_in[2];
  const float* w_f32[8] = {
      (const float*)d_in[3], (const float*)d_in[4], (const float*)d_in[5],
      (const float*)d_in[6], (const float*)d_in[7], (const float*)d_in[8],
      (const float*)d_in[9], (const float*)d_in[10]};
  const int w_n[8] = {DM * DC, DC * 2048, DM * 1024, DM * DC,
                      DC * 2048, DC * 2048, DM * 1024, 2048 * DM};

  size_t off = 0;
  auto alloc = [&](size_t bytes) {
    void* p = (char*)d_ws + off;
    off += (bytes + 255) & ~(size_t)255;
    return p;
  };
  u16* x_bf = (u16*)alloc((size_t)M * DM * 2);
  u16* w_bf[8];
  for (int i = 0; i < 8; ++i) w_bf[i] = (u16*)alloc((size_t)w_n[i] * 2);
  u16* c_q  = (u16*)alloc((size_t)M * DC * 2);
  u16* c_kv = (u16*)alloc((size_t)M * DC * 2);
  u16* qbuf = (u16*)alloc((size_t)M * H * 192 * 2);
  u16* kbuf = (u16*)alloc((size_t)M * H * 192 * 2);
  u16* vbuf = (u16*)alloc((size_t)M * 2048 * 2);
  u16* attn = (u16*)alloc((size_t)M * 2048 * 2);

  // fp32 -> bf16 conversions
  {
    int n = M * DM;
    cvt_f32_bf16<<<(n + 255) / 256, 256, 0, stream>>>(x, x_bf, n);
    for (int i = 0; i < 8; ++i)
      cvt_f32_bf16<<<(w_n[i] + 255) / 256, 256, 0, stream>>>(w_f32[i], w_bf[i], w_n[i]);
  }

  auto G = [](int N_, int M_) { return dim3(N_ / 64, M_ / 64); };

  // latent projections
  gemm_bf16<true><<<G(512, M), 256, 0, stream>>>(x_bf, w_bf[0], c_q,  M, 512, DM, 512, 0, 0);
  gemm_bf16<true><<<G(512, M), 256, 0, stream>>>(x_bf, w_bf[3], c_kv, M, 512, DM, 512, 0, 0);
  // up-projections written interleaved into [tok, H, 192]
  gemm_bf16<true><<<G(2048, M), 256, 0, stream>>>(c_q,  w_bf[1], qbuf, M, 2048, DC, 3072, 128, 0);
  gemm_bf16<true><<<G(1024, M), 256, 0, stream>>>(x_bf, w_bf[2], qbuf, M, 1024, DM, 3072, 64, 128);
  gemm_bf16<true><<<G(2048, M), 256, 0, stream>>>(c_kv, w_bf[4], kbuf, M, 2048, DC, 3072, 128, 0);
  gemm_bf16<true><<<G(1024, M), 256, 0, stream>>>(x_bf, w_bf[6], kbuf, M, 1024, DM, 3072, 64, 128);
  gemm_bf16<true><<<G(2048, M), 256, 0, stream>>>(c_kv, w_bf[5], vbuf, M, 2048, DC, 2048, 0, 0);

  // rotary on rope columns of q and k
  rope_inplace<<<M * H, 32, 0, stream>>>(qbuf, fcos, fsin, S, H);
  rope_inplace<<<M * H, 32, 0, stream>>>(kbuf, fcos, fsin, S, H);

  // causal flash attention: B*H*(S/64) = 1024 blocks
  mla_attention<<<B * H * (S / 64), 256, 0, stream>>>(qbuf, kbuf, vbuf, attn, S);

  // output projection -> fp32 d_out
  gemm_bf16<false><<<G(2048, M), 256, 0, stream>>>(attn, w_bf[7], (float*)d_out,
                                                   M, 2048, 2048, 2048, 0, 0);
}
